// SpatialCrossAttention_13606456393868
// MI455X (gfx1250) — matrix operand order repaired and checked by reading, not compiled, not run on hardware
//
#include <hip/hip_runtime.h>
#include <hip/hip_bf16.h>
#include <cstdint>

// ---------------- problem constants (fixed by the reference) ----------------
#define EMBED    256
#define HEADS    8
#define HEAD_DIM 32
#define LEVELS   4
#define POINTS   8
#define CAMS     6
#define NQ       2500
#define NV       19560
#define KDIM     256   // all GEMMs have K = EMBED = 256
#define KSTEPS   (KDIM / 32)

constexpr int LVL_H[4] = {92, 46, 23, 12};
constexpr int LVL_W[4] = {160, 80, 40, 20};
constexpr int LVL_S[4] = {0, 14720, 18400, 19320};

// ---------------- bf16 helpers (bit ops only, RNE) ----------------
__device__ __forceinline__ unsigned short f2bf(float f) {
  unsigned int b = __builtin_bit_cast(unsigned int, f);
  unsigned int r = b + 0x7FFFu + ((b >> 16) & 1u);
  return (unsigned short)(r >> 16);
}
__device__ __forceinline__ unsigned int pack2bf(float lo, float hi) {
  return (unsigned int)f2bf(lo) | ((unsigned int)f2bf(hi) << 16);
}
__device__ __forceinline__ float bf2f(unsigned short u) {
  unsigned int b = ((unsigned int)u) << 16;
  return __builtin_bit_cast(float, b);
}

// ---------------- CDNA5 async global->LDS copy (ASYNCcnt tracked) ----------
__device__ __forceinline__ void async_g2l_b128(unsigned lds_off,
                                               const void* gaddr) {
  asm volatile("global_load_async_to_lds_b128 %0, %1, off"
               :
               : "v"(lds_off), "v"(gaddr)
               : "memory");
}
template <int N>
__device__ __forceinline__ void wait_asynccnt() {
  if constexpr (N == 0)
    asm volatile("s_wait_asynccnt 0x0" ::: "memory");
  else if constexpr (N == 2)
    asm volatile("s_wait_asynccnt 0x2" ::: "memory");
  else
    asm volatile("s_wait_asynccnt 0x3" ::: "memory");
}
// Generic-space shared pointer -> 32-bit LDS byte address (aperture strip).
__device__ __forceinline__ unsigned lds_addr_of(const void* p) {
  return (unsigned)(uintptr_t)p;
}

// ---------------- WMMA types ----------------
typedef __attribute__((ext_vector_type(16))) __bf16        v16bf;
typedef __attribute__((ext_vector_type(8)))  float         v8f;
typedef __attribute__((ext_vector_type(8)))  unsigned int  v8u;

// LDS tile strides in u32 (bf16 pairs); 16B-multiple rows for async b128.
#define AS_STRIDE 20   // A: 16 data pairs + 4 pad  (80B rows)
#define BS_STRIDE 72   // B: 64 data pairs + 8 pad  (288B rows)

// ============================================================================
// f32 -> bf16 pair-packing convert (for weights); n2 = pairs.
// ============================================================================
__global__ __launch_bounds__(256) void cvt_f32_bf16_kernel(
    const float* __restrict__ in, unsigned int* __restrict__ out, int n2) {
  const int i = blockIdx.x * 256 + threadIdx.x;
  if (i < n2) {
    const float2 f = *(const float2*)(in + (size_t)i * 2);
    out[i] = pack2bf(f.x, f.y);
  }
}

// ============================================================================
// Generic K=256 GEMM:  C(M,N) = A(M,256) * B(256,N) + bias [+ residual]
// B is bf16 (pre-converted weights), double-buffered async-LDS staging.
// AMODE: 0 = A is f32 (sync b128 loads + convert while staging)
//        1 = A is bf16 (async b128 chunks)
// OMODE: 0 = store f32, 1 = store bf16, 2 = store f32 + residual (f32)
// Block: 256 threads = 8 waves as 2(M)x4(N); block tile 64x128.
// Each wave: 32x32 C tile = 2 A-frags x 2 B-frags = 4 v_wmma per K-step.
// Pipeline: issue async copy of tile k+1, s_wait_asynccnt <= N (in-order
// completion => tile k landed), barrier, compute tile k.
// ============================================================================
template <int AMODE, int OMODE>
__global__ __launch_bounds__(256) void gemm256_wmma(
    const void* __restrict__ Aq, const unsigned short* __restrict__ B,
    const float* __restrict__ bias, const float* __restrict__ residual,
    void* __restrict__ C, int M, int N) {
  __shared__ unsigned int As[2][64 * AS_STRIDE];
  __shared__ unsigned int Bs[2][32 * BS_STRIDE];

  const int t    = threadIdx.x;
  const int lane = t & 31;
  const int wid  = t >> 5;
  const int wm   = wid & 1;   // 0..1  (M sub-tile of 32 rows)
  const int wn   = wid >> 1;  // 0..3  (N sub-tile of 32 cols)
  const int m0   = blockIdx.x * 64;
  const int n0   = blockIdx.y * 128;

  v8f acc[2][2] = {{{}, {}}, {{}, {}}};

  // ---- tile staging (A: 64x32, B: 32x128) into buffer `buf` ----
  auto stage = [&](int k0, int buf) {
    if (AMODE == 0) {
      // A f32: one row-octet per thread: 2x global b128 + convert + ds b128
      const int row = t >> 2;        // 0..63
      const int cg  = t & 3;         // 4 chunks of 8 floats per 32-wide row
      int gr = m0 + row;
      if (gr >= M) gr = M - 1;       // clamp (stores guarded later)
      const float* g = (const float*)Aq + (size_t)gr * KDIM + k0 + cg * 8;
      const float4 f0 = *(const float4*)(g + 0);
      const float4 f1 = *(const float4*)(g + 4);
      uint4 u;
      u.x = pack2bf(f0.x, f0.y);
      u.y = pack2bf(f0.z, f0.w);
      u.z = pack2bf(f1.x, f1.y);
      u.w = pack2bf(f1.z, f1.w);
      *(uint4*)&As[buf][row * AS_STRIDE + cg * 4] = u;
    } else {
      // A bf16: async DMA, one 16B chunk per thread (64 rows x 4 chunks)
      const int row   = t >> 2;
      const int chunk = t & 3;
      int gr = m0 + row;
      if (gr >= M) gr = M - 1;
      const unsigned short* g =
          (const unsigned short*)Aq + (size_t)gr * KDIM + k0 + chunk * 8;
      async_g2l_b128(lds_addr_of(&As[buf][0]) +
                         (unsigned)(row * (AS_STRIDE * 4) + chunk * 16),
                     g);
    }
    // B bf16: async DMA, 32 rows x 16 chunks of 16B = 512 chunks, 2/thread
    const unsigned bbase = lds_addr_of(&Bs[buf][0]);
#pragma unroll
    for (int i = 0; i < 2; ++i) {
      const int idx   = t + i * 256;  // 0..511
      const int row   = idx >> 4;     // 0..31
      const int chunk = idx & 15;     // 16B chunks across 256B row
      const unsigned short* g = B + (size_t)(k0 + row) * N + n0 + chunk * 8;
      async_g2l_b128(bbase + (unsigned)(row * (BS_STRIDE * 4) + chunk * 16), g);
    }
  };
  // async ops issued per thread per stage (for the pipelined wait)
  constexpr int NOPS = (AMODE == 0) ? 2 : 3;

  stage(0, 0);
  for (int it = 0; it < KSTEPS; ++it) {
    const int cur = it & 1;
    if (it + 1 < KSTEPS) {
      stage((it + 1) * 32, cur ^ 1);   // overlap next tile's DMA with compute
      // prefetch the tile after that into L2 (global_prefetch_b8)
      if (it + 2 < KSTEPS) {
        const size_t esz = (AMODE == 0) ? 4 : 2;
        int pr = m0 + (t & 63);
        if (pr >= M) pr = M - 1;
        __builtin_prefetch(
            (const char*)Aq + ((size_t)pr * KDIM + (it + 2) * 32) * esz, 0, 1);
        __builtin_prefetch(
            (const char*)(B + (size_t)((it + 2) * 32 + (t & 31)) * N + n0), 0, 1);
      }
      wait_asynccnt<NOPS>();           // current tile done; next still in flight
    } else {
      wait_asynccnt<0>();
    }
    __syncthreads();                    // all threads' tile data visible

    // ---- A fragments (16x32 bf16): lane = {M row, K half} per ISA layout ----
    v16bf a[2];
#pragma unroll
    for (int mi = 0; mi < 2; ++mi) {
      const int arow = wm * 32 + mi * 16 + (lane & 15);
      const int base = (lane >> 4) * 4;  // K-pair base: 0 / 4
      const unsigned int* ap = &As[cur][arow * AS_STRIDE];
      v8u au;
#pragma unroll
      for (int v = 0; v < 4; ++v) au[v] = ap[base + v];         // K 0..7 / 8..15
#pragma unroll
      for (int v = 0; v < 4; ++v) au[4 + v] = ap[8 + base + v]; // K 16..23 / 24..31
      a[mi] = __builtin_bit_cast(v16bf, au);
    }
    // ---- B fragments (32x16 bf16): lane = N col, elements = K (same order as A) ----
    const unsigned short* bsh = (const unsigned short*)&Bs[cur][0];
    const int khalf = (lane >> 4) * 8;
    v16bf b[2];
#pragma unroll
    for (int ni = 0; ni < 2; ++ni) {
      const int col = wn * 32 + ni * 16 + (lane & 15);
#pragma unroll
      for (int s = 0; s < 16; ++s) {
        const int k = (s < 8) ? (khalf + s) : (16 + khalf + (s - 8));
        b[ni][s] = __builtin_bit_cast(__bf16, bsh[k * (BS_STRIDE * 2) + col]);
      }
    }
#pragma unroll
    for (int mi = 0; mi < 2; ++mi)
#pragma unroll
      for (int ni = 0; ni < 2; ++ni)
        acc[mi][ni] = __builtin_amdgcn_wmma_f32_16x16x32_bf16(
            false, a[mi], false, b[ni], (short)0, acc[mi][ni], false, false);

    __syncthreads();  // reads done before next stage overwrites this buffer
  }

  // ---- epilogue: C layout VGPR r -> M = r + 8*(lane>=16), N = lane&15 ----
  const int rbase = (lane >> 4) * 8;
#pragma unroll
  for (int mi = 0; mi < 2; ++mi) {
#pragma unroll
    for (int r = 0; r < 8; ++r) {
      const int grow = m0 + wm * 32 + mi * 16 + rbase + r;
      if (grow < M) {
#pragma unroll
        for (int ni = 0; ni < 2; ++ni) {
          const int gcol = n0 + wn * 32 + ni * 16 + (lane & 15);
          float v = acc[mi][ni][r] + bias[gcol];
          if (OMODE == 0)
            ((float*)C)[(size_t)grow * N + gcol] = v;
          else if (OMODE == 1)
            ((unsigned short*)C)[(size_t)grow * N + gcol] = f2bf(v);
          else
            ((float*)C)[(size_t)grow * N + gcol] =
                v + residual[(size_t)grow * N + gcol];
        }
      }
    }
  }
}

// ============================================================================
// Softmax over 32 (LEVELS*POINTS) per (query, head); wave32 = one (q,h).
// In-place on att (NQ, HEADS, 32).
// ============================================================================
__global__ __launch_bounds__(256) void softmax32_kernel(float* __restrict__ att) {
  const int q    = blockIdx.x;
  const int lane = threadIdx.x & 31;
  const int h    = threadIdx.x >> 5;
  const size_t idx = (size_t)q * (HEADS * 32) + h * 32 + lane;
  float v  = att[idx];
  float mx = v;
#pragma unroll
  for (int s = 16; s; s >>= 1) mx = fmaxf(mx, __shfl_xor(mx, s, 32));
  const float e = __expf(v - mx);
  float sum = e;
#pragma unroll
  for (int s = 16; s; s >>= 1) sum += __shfl_xor(sum, s, 32);
  att[idx] = e / sum;
}

// ============================================================================
// Deformable bilinear sampling. Block = (q, cam); wave = head; lane = dim.
// vproj is bf16 (CAMS, NV, EMBED) -> every corner gather is a coalesced 64B line.
// ============================================================================
__global__ __launch_bounds__(256) void sample_kernel(
    const unsigned short* __restrict__ vproj, const float* __restrict__ off,
    const float* __restrict__ attw, const float* __restrict__ refpts,
    float* __restrict__ out_cam) {
  const int q    = blockIdx.x;
  const int c    = blockIdx.y;
  const int lane = threadIdx.x & 31;
  const int h    = threadIdx.x >> 5;

  const float* offq = off + (size_t)q * (HEADS * LEVELS * POINTS * 2) +
                      h * (LEVELS * POINTS * 2);
  const float* awq  = attw + (size_t)q * (HEADS * LEVELS * POINTS) +
                      h * (LEVELS * POINTS);
  const float* refq = refpts + ((size_t)c * NQ + q) * 8;  // (D=4, 2)

  float acc = 0.f;
#pragma unroll
  for (int lvl = 0; lvl < LEVELS; ++lvl) {
    const int W = LVL_W[lvl], H = LVL_H[lvl];
    const float fW = (float)W, fH = (float)H;
    const size_t vbase =
        ((size_t)c * NV + LVL_S[lvl]) * EMBED + h * HEAD_DIM + lane;
    for (int p = 0; p < POINTS; ++p) {
      const int d  = p & 3;  // POINTS//D groups of D reference points
      const float rx = refq[d * 2 + 0];
      const float ry = refq[d * 2 + 1];
      const float ox = offq[(lvl * POINTS + p) * 2 + 0];
      const float oy = offq[(lvl * POINTS + p) * 2 + 1];
      const float aw = awq[lvl * POINTS + p];
      // loc = ref + off/norm;  x = loc.x*W - 0.5  ==  ref.x*W + off.x - 0.5
      const float x  = rx * fW + ox - 0.5f;
      const float y  = ry * fH + oy - 0.5f;
      const float xf = floorf(x), yf = floorf(y);
      const float lx = x - xf, ly = y - yf;
      const int x0 = (int)xf, y0 = (int)yf;
      float s = 0.f;
#pragma unroll
      for (int cy = 0; cy < 2; ++cy) {
        const int yi = y0 + cy;
        if (yi < 0 || yi >= H) continue;
        const float wy = cy ? ly : (1.f - ly);
#pragma unroll
        for (int cx = 0; cx < 2; ++cx) {
          const int xi = x0 + cx;
          if (xi < 0 || xi >= W) continue;
          const float wx = cx ? lx : (1.f - lx);
          const float vv = bf2f(vproj[vbase + (size_t)(yi * W + xi) * EMBED]);
          s = fmaf(wx * wy, vv, s);
        }
      }
      acc = fmaf(aw, s, acc);
    }
  }
  out_cam[((size_t)c * NQ + q) * EMBED + h * HEAD_DIM + lane] = acc;
}

// ============================================================================
// Masked combine over cameras -> bf16 slots for the output GEMM.
// ============================================================================
__global__ __launch_bounds__(256) void combine_kernel(
    const float* __restrict__ out_cam, const unsigned char* __restrict__ bev_mask,
    unsigned short* __restrict__ slots_bf16) {
  const int q = blockIdx.x;
  const int e = threadIdx.x;
  float sum = 0.f;
  int count = 0;
#pragma unroll
  for (int c = 0; c < CAMS; ++c) {
    // bev_mask layout (CAMS, 1, NQ, D=4) of bool bytes; 4B-aligned read
    const unsigned int mm =
        *(const unsigned int*)(bev_mask + ((size_t)c * NQ + q) * 4);
    if (mm != 0u) {
      ++count;
      sum += out_cam[((size_t)c * NQ + q) * EMBED + e];
    }
  }
  const float cnt = (float)(count > 0 ? count : 1);
  slots_bf16[(size_t)q * EMBED + e] = f2bf(sum / cnt);
}

// ============================================================================
// Host-side launcher
// ============================================================================
extern "C" void kernel_launch(void* const* d_in, const int* in_sizes, int n_in,
                              void* d_out, int out_size, void* d_ws,
                              size_t ws_size, hipStream_t stream) {
  (void)in_sizes; (void)n_in; (void)out_size; (void)ws_size;
  const float*         query    = (const float*)d_in[0];
  // d_in[1] = key (unused by the reference computation)
  const float*         value    = (const float*)d_in[2];
  const float*         refpts   = (const float*)d_in[3];
  const unsigned char* bev_mask = (const unsigned char*)d_in[4];
  // d_in[5] spatial_shapes, d_in[6] level_start_index: compile-time constants
  const float* Wv   = (const float*)d_in[7];
  const float* bv   = (const float*)d_in[8];
  const float* Woff = (const float*)d_in[9];
  const float* boff = (const float*)d_in[10];
  const float* Watt = (const float*)d_in[11];
  const float* batt = (const float*)d_in[12];
  const float* Wout = (const float*)d_in[13];
  const float* bout = (const float*)d_in[14];
  float* out = (float*)d_out;

  // ---- scratch carve-out (~86 MB) ----
  char*  ws    = (char*)d_ws;
  size_t wsoff = 0;
  auto take = [&](size_t bytes) -> char* {
    char* p = ws + wsoff;
    wsoff   = (wsoff + bytes + 255) & ~(size_t)255;
    return p;
  };
  unsigned short* vproj   = (unsigned short*)take((size_t)CAMS * NV * EMBED * 2);
  float*          off_raw = (float*)take((size_t)NQ * 512 * 4);
  float*          att     = (float*)take((size_t)NQ * 256 * 4);
  float*          out_cam = (float*)take((size_t)CAMS * NQ * EMBED * 4);
  unsigned short* slots   = (unsigned short*)take((size_t)NQ * EMBED * 2);
  unsigned short* Wv_bf   = (unsigned short*)take((size_t)KDIM * EMBED * 2);
  unsigned short* Woff_bf = (unsigned short*)take((size_t)KDIM * 512 * 2);
  unsigned short* Watt_bf = (unsigned short*)take((size_t)KDIM * 256 * 2);
  unsigned short* Wout_bf = (unsigned short*)take((size_t)KDIM * EMBED * 2);

  const dim3 blk(256);
  const int  MV = CAMS * NV;  // 117360 rows

  // 0) convert weight matrices to bf16 (tiny; enables async-LDS B staging)
  {
    const int nWv = KDIM * EMBED / 2, nWo = KDIM * 512 / 2;
    cvt_f32_bf16_kernel<<<(nWv + 255) / 256, blk, 0, stream>>>(
        Wv, (unsigned int*)Wv_bf, nWv);
    cvt_f32_bf16_kernel<<<(nWo + 255) / 256, blk, 0, stream>>>(
        Woff, (unsigned int*)Woff_bf, nWo);
    cvt_f32_bf16_kernel<<<(nWv + 255) / 256, blk, 0, stream>>>(
        Watt, (unsigned int*)Watt_bf, nWv);
    cvt_f32_bf16_kernel<<<(nWv + 255) / 256, blk, 0, stream>>>(
        Wout, (unsigned int*)Wout_bf, nWv);
  }

  // 1) value projection -> bf16 vproj (dominant GEMM; L2-resident afterwards)
  gemm256_wmma<0, 1><<<dim3((MV + 63) / 64, EMBED / 128), blk, 0, stream>>>(
      value, Wv_bf, bv, nullptr, vproj, MV, EMBED);
  // 2) sampling offsets (query is cam-invariant -> computed once)
  gemm256_wmma<0, 0><<<dim3((NQ + 63) / 64, 512 / 128), blk, 0, stream>>>(
      query, Woff_bf, boff, nullptr, off_raw, NQ, 512);
  // 3) attention logits
  gemm256_wmma<0, 0><<<dim3((NQ + 63) / 64, 256 / 128), blk, 0, stream>>>(
      query, Watt_bf, batt, nullptr, att, NQ, 256);
  // 4) softmax over the 32 (level,point) weights per (q, head)
  softmax32_kernel<<<NQ, blk, 0, stream>>>(att);
  // 5) deformable bilinear sampling per camera
  sample_kernel<<<dim3(NQ, CAMS), blk, 0, stream>>>(vproj, off_raw, att, refpts,
                                                    out_cam);
  // 6) masked combine over cameras
  combine_kernel<<<NQ, blk, 0, stream>>>(out_cam, bev_mask, slots);
  // 7) output projection + bias + residual query -> d_out (f32)
  gemm256_wmma<1, 2><<<dim3((NQ + 63) / 64, EMBED / 128), blk, 0, stream>>>(
      slots, Wout_bf, bout, query, out, NQ, EMBED);
}
